// PhotonicQuantumAttention_9045201125569
// MI455X (gfx1250) — matrix operand order repaired and checked by reading, not compiled, not run on hardware
//
#include <hip/hip_runtime.h>
#include <hip/hip_bf16.h>
#include <math.h>

// ---------------------------------------------------------------------------
// Types for CDNA5 WMMA (wave32): V_WMMA_F32_16X16X32_BF16
// ---------------------------------------------------------------------------
typedef __attribute__((ext_vector_type(16))) __bf16 v16bf;
typedef __attribute__((ext_vector_type(8)))  float  v8f;

#define EMBED_DIM 1024
#define NUM_HEADS 8
#define HEAD_DIM  128
#define CIRCUITS  4
#define DEPTH     3
#define WAVELEN   16
#define SCALING   0.08838834764831845f  // 128^-0.5
#define TM 128
#define TN 64
#define TK 32

// round-to-nearest-even fp32 -> bf16 (hi) plus bf16 residual (lo)
__device__ inline void f32_split_bf16(__bf16* hi, __bf16* lo, float f) {
  unsigned u  = __float_as_uint(f);
  unsigned rh = (u + 0x7FFFu + ((u >> 16) & 1u)) >> 16;
  unsigned short hs = (unsigned short)rh;
  float fh = __uint_as_float(((unsigned)hs) << 16);
  float r  = f - fh;
  unsigned ur = __float_as_uint(r);
  unsigned rl = (ur + 0x7FFFu + ((ur >> 16) & 1u)) >> 16;
  *hi = __builtin_bit_cast(__bf16, hs);
  *lo = __builtin_bit_cast(__bf16, (unsigned short)rl);
}

__device__ inline float wave_sum32(float v) {
  #pragma unroll
  for (int o = 16; o > 0; o >>= 1) v += __shfl_xor(v, o, 32);
  return v;
}

// CDNA5 async DMA: global -> LDS, 16 bytes per lane, tracked by ASYNCcnt.
__device__ inline void async_load_b128(unsigned lds_off, const float* gaddr) {
  asm volatile("global_load_async_to_lds_b128 %0, %1, off"
               :: "v"(lds_off), "v"(gaddr) : "memory");
}
__device__ inline unsigned lds_off32(const void* p) {
  // On AMDGPU the generic (flat) address of LDS carries the LDS byte offset
  // in its low 32 bits (shared aperture in the high bits).
  return (unsigned)(uintptr_t)p;
}

// ---------------------------------------------------------------------------
// GEMM: C[M,N] = A[M,K] * B[N,K]^T + bias[N]
// fp32 in/out; bf16x3 split WMMA core; double-buffered LDS tiles filled by
// GLOBAL_LOAD_ASYNC_TO_LDS_B128 (ASYNCcnt) so DMA overlaps the WMMA burst.
// Block: 256 threads = 8 waves. Tile: 128(M) x 64(N), K-step 32.
// ---------------------------------------------------------------------------
__global__ __launch_bounds__(256) void wmma_gemm_bf16x3_async(
    const float* __restrict__ A, const float* __restrict__ B,
    const float* __restrict__ bias, float* __restrict__ C,
    int M, int N, int K)
{
  __shared__ float Asf[2][TM * TK];   // 2 x 16 KB
  __shared__ float Bsf[2][TN * TK];   // 2 x  8 KB

  const int tid  = threadIdx.x;
  const int wave = tid >> 5;
  const int lane = tid & 31;
  const int lh   = lane & 15;
  const int half = lane >> 4;          // 0: lanes 0-15, 1: lanes 16-31
  const int m0   = wave * 16;          // wave's M offset inside the tile
  const int gm0  = blockIdx.y * TM;
  const int gn0  = blockIdx.x * TN;

  // Per-thread staging map: 4 A-chunks + 2 B-chunks of float4 per tile.
  const int a_m[4] = { (tid + 0)   >> 3, (tid + 256) >> 3,
                       (tid + 512) >> 3, (tid + 768) >> 3 };
  const int a_c   = (tid & 7) * 4;               // same c4 for all (tid+256k share low bits)
  const int b_n[2] = { (tid + 0) >> 3, (tid + 256) >> 3 };

  auto issue_tile = [&](int buf, int k0) {
    #pragma unroll
    for (int i = 0; i < 4; ++i) {
      const float* g = A + (size_t)(gm0 + a_m[i]) * K + k0 + a_c;
      async_load_b128(lds_off32(&Asf[buf][a_m[i] * TK + a_c]), g);
    }
    #pragma unroll
    for (int i = 0; i < 2; ++i) {
      const float* g = B + (size_t)(gn0 + b_n[i]) * K + k0 + a_c;
      async_load_b128(lds_off32(&Bsf[buf][b_n[i] * TK + a_c]), g);
    }
  };

  v8f acc[4] = {};

  issue_tile(0, 0);                    // prologue DMA for first K tile
  int cur = 0;

  for (int k0 = 0; k0 < K; k0 += TK) {
    const bool have_next = (k0 + TK) < K;
    if (have_next) {
      __syncthreads();                 // nobody still reading buf cur^1
      issue_tile(cur ^ 1, k0 + TK);    // overlap next tile DMA with compute
      // current tile's 6 DMAs done; next tile's 6 still in flight
      asm volatile("s_wait_asynccnt 0x6" ::: "memory");
      if (k0 + 2 * TK < K) {           // warm L2 two tiles ahead
        __builtin_prefetch(A + (size_t)(gm0 + (tid >> 1)) * K + k0 + 2 * TK, 0, 1);
        __builtin_prefetch(B + (size_t)(gn0 + (tid & 63)) * K + k0 + 2 * TK, 0, 1);
      }
    } else {
      asm volatile("s_wait_asynccnt 0x0" ::: "memory");
    }
    __syncthreads();                   // cross-wave visibility of tile `cur`

    // A fragment (16x32): lane<16 -> K 0..15,
    // lane>=16 -> K 16..31; row M = m0 + (lane&15). Split fp32->bf16 here.
    v16bf afh, afl;
    {
      const float* src = &Asf[cur][(m0 + lh) * TK];
      const int kb0 = half * 16;
      const int kb1 = half * 16 + 8;
      #pragma unroll
      for (int i = 0; i < 8; ++i) {
        __bf16 h, l;
        f32_split_bf16(&h, &l, src[kb0 + i]);  afh[i] = h;      afl[i] = l;
        f32_split_bf16(&h, &l, src[kb1 + i]);  afh[8 + i] = h;  afl[8 + i] = l;
      }
    }
    #pragma unroll
    for (int nt = 0; nt < 4; ++nt) {
      // B fragment (32x16): lane<16 -> K 0..15, lane>=16 -> K 16..31, col N = lane&15
      v16bf bfh, bfl;
      const float* src = &Bsf[cur][(nt * 16 + lh) * TK + half * 16];
      #pragma unroll
      for (int i = 0; i < 16; ++i) {
        __bf16 h, l;
        f32_split_bf16(&h, &l, src[i]);
        bfh[i] = h;  bfl[i] = l;
      }
      // bf16x3 split product: Ah*Bh + Ah*Bl + Al*Bh  (near-fp32 accuracy)
      acc[nt] = __builtin_amdgcn_wmma_f32_16x16x32_bf16(false, afh, false, bfh,
                                                        (short)0, acc[nt], false, false);
      acc[nt] = __builtin_amdgcn_wmma_f32_16x16x32_bf16(false, afh, false, bfl,
                                                        (short)0, acc[nt], false, false);
      acc[nt] = __builtin_amdgcn_wmma_f32_16x16x32_bf16(false, afl, false, bfh,
                                                        (short)0, acc[nt], false, false);
    }
    cur ^= 1;
  }

  // Epilogue per ISA 32-bit C/D layout: VGPR r -> M = r + 8*half, N = lane&15
  #pragma unroll
  for (int nt = 0; nt < 4; ++nt) {
    const int n  = gn0 + nt * 16 + lh;
    const float bv = bias[n];
    #pragma unroll
    for (int r = 0; r < 8; ++r) {
      const int m = gm0 + m0 + half * 8 + r;
      C[(size_t)m * N + n] = acc[nt][r] + bv;
    }
  }
}

// ---------------------------------------------------------------------------
// Middle stage: one block per token t (B*S = 4096 tokens), 8 waves = 8 heads.
// Wave h owns head h's 128 dims (4 values per lane).
// ---------------------------------------------------------------------------
__global__ __launch_bounds__(256) void quantum_middle(
    const float* __restrict__ qkv,      // [4096][3072]
    const float* __restrict__ phases,   // [16][1024]
    const float* __restrict__ gates,    // [3][8][4]
    const float* __restrict__ qw,       // [4][8][8]
    const float* __restrict__ logits,   // [4]
    float* __restrict__ measured)       // [4096][1024]
{
  __shared__ float qre[NUM_HEADS][HEAD_DIM], qim[NUM_HEADS][HEAD_DIM];
  __shared__ float kre[NUM_HEADS][HEAD_DIM], kim[NUM_HEADS][HEAD_DIM];
  __shared__ float vre[NUM_HEADS][HEAD_DIM], vim[NUM_HEADS][HEAD_DIM];
  __shared__ float cohr[HEAD_DIM], cohi[HEAD_DIM];
  __shared__ float patr[64], pati[64], sb[64];

  const int t    = blockIdx.x;
  const int tid  = threadIdx.x;
  const int h    = tid >> 5;
  const int lane = tid & 31;
  const float* base = qkv + (size_t)t * 3072;

  // ---- quantum encode (per-head L2 normalize + photonic phase) ----
  float qv[4], kv[4], vv[4];
  float sq = 0.f, sk = 0.f, sv = 0.f;
  #pragma unroll
  for (int i = 0; i < 4; ++i) {
    const int d = lane + 32 * i;
    qv[i] = base[h * HEAD_DIM + d];
    kv[i] = base[1024 + h * HEAD_DIM + d];
    vv[i] = base[2048 + h * HEAD_DIM + d];
    sq += qv[i] * qv[i];  sk += kv[i] * kv[i];  sv += vv[i] * vv[i];
  }
  sq = wave_sum32(sq); sk = wave_sum32(sk); sv = wave_sum32(sv);
  const float iq = 1.f / fmaxf(sqrtf(sq), 1e-12f);
  const float ik = 1.f / fmaxf(sqrtf(sk), 1e-12f);
  const float iv = 1.f / fmaxf(sqrtf(sv), 1e-12f);
  #pragma unroll
  for (int i = 0; i < 4; ++i) {
    const int d = lane + 32 * i;
    const float ph = phases[(d & (WAVELEN - 1)) * EMBED_DIM + d];  // [d%16, d]
    const float cp = cosf(ph), sp = sinf(ph);
    const float qn = qv[i] * iq, kn = kv[i] * ik, vn = vv[i] * iv;
    qre[h][d] = qn * cp;  qim[h][d] = qn * sp;
    kre[h][d] = kn * cp;  kim[h][d] = kn * sp;
    vre[h][d] = vn * cp;  vim[h][d] = vn * sp;
  }
  __syncthreads();

  // ---- entangle q and k across adjacent heads (3 depths) ----
  for (int depth = 0; depth < DEPTH; ++depth) {
    float nqr[4], nqi[4], nkr[4], nki[4];
    const bool active = (h < NUM_HEADS - 1);
    if (active) {
      const float th = gates[depth * 32 + h * 4 + 0];
      const float fi = gates[depth * 32 + h * 4 + 1];
      const float ct = cosf(0.5f * th), st = sinf(0.5f * th);
      const float cf = cosf(fi),        sf = sinf(fi);
      #pragma unroll
      for (int i = 0; i < 4; ++i) {
        const int d = lane + 32 * i;
        // q state: new[h] = |s_h|^2 * rot(s_{h+1}) + (1-|s_h|^2) * s_{h+1}
        float s1r = qre[h][d],     s1i = qim[h][d];
        float s2r = qre[h + 1][d], s2i = qim[h + 1][d];
        float ctrl = s1r * s1r + s1i * s1i;
        float rr = ct * s2r + st * (cf * s2r + sf * s2i);
        float ri = ct * s2i + st * (sf * s2r - cf * s2i);
        nqr[i] = ctrl * rr + (1.f - ctrl) * s2r;
        nqi[i] = ctrl * ri + (1.f - ctrl) * s2i;
        // k state
        s1r = kre[h][d];     s1i = kim[h][d];
        s2r = kre[h + 1][d]; s2i = kim[h + 1][d];
        ctrl = s1r * s1r + s1i * s1i;
        rr = ct * s2r + st * (cf * s2r + sf * s2i);
        ri = ct * s2i + st * (sf * s2r - cf * s2i);
        nkr[i] = ctrl * rr + (1.f - ctrl) * s2r;
        nki[i] = ctrl * ri + (1.f - ctrl) * s2i;
      }
    }
    __syncthreads();
    if (active) {
      #pragma unroll
      for (int i = 0; i < 4; ++i) {
        const int d = lane + 32 * i;
        qre[h][d] = nqr[i];  qim[h][d] = nqi[i];
        kre[h][d] = nkr[i];  kim[h][d] = nki[i];
      }
    }
    __syncthreads();
  }

  // ---- coherence phase from head-sum of vq: e^{i*angle(sum_h vq)} ----
  if (tid < HEAD_DIM) {
    float sr = 0.f, si = 0.f;
    #pragma unroll
    for (int hh = 0; hh < NUM_HEADS; ++hh) { sr += vre[hh][tid]; si += vim[hh][tid]; }
    const float mag = sqrtf(sr * sr + si * si);
    cohr[tid] = (mag > 0.f) ? sr / mag : 1.f;   // angle(0) = 0 -> e^{i0} = 1
    cohi[tid] = (mag > 0.f) ? si / mag : 0.f;
  }

  // ---- base pattern: pat[i][j] = sum_d conj(q_ent[i,d]) * k_ent[j,d] ----
  #pragma unroll
  for (int j = 0; j < NUM_HEADS; ++j) {
    float pr = 0.f, pi = 0.f;
    #pragma unroll
    for (int i4 = 0; i4 < 4; ++i4) {
      const int d = lane + 32 * i4;
      const float ar = qre[h][d], ai = qim[h][d];
      const float br = kre[j][d], bi = kim[j][d];
      pr += ar * br + ai * bi;   // Re(conj(a)*b)
      pi += ar * bi - ai * br;   // Im(conj(a)*b)
    }
    pr = wave_sum32(pr);
    pi = wave_sum32(pi);
    if (lane == 0) { patr[h * 8 + j] = pr; pati[h * 8 + j] = pi; }
  }
  __syncthreads();
  // |pattern|^2 * SCALING (per-circuit phase mod cancels inside |.|^2)
  if (tid < 64) {
    const float pr = patr[tid], pi = pati[tid];
    sb[tid] = (pr * pr + pi * pi) * SCALING;
  }
  __syncthreads();

  // ---- superposition weights ----
  float sup[CIRCUITS];
  {
    const float l0 = logits[0], l1 = logits[1], l2 = logits[2], l3 = logits[3];
    const float mx = fmaxf(fmaxf(l0, l1), fmaxf(l2, l3));
    sup[0] = expf(l0 - mx); sup[1] = expf(l1 - mx);
    sup[2] = expf(l2 - mx); sup[3] = expf(l3 - mx);
    const float z = sup[0] + sup[1] + sup[2] + sup[3];
    sup[0] /= z; sup[1] /= z; sup[2] /= z; sup[3] /= z;
  }

  // ---- circuit loop: scaled softmax over heads, weighted vq, clamp+combine ----
  float accf[4] = {0.f, 0.f, 0.f, 0.f};
  for (int c = 0; c < CIRCUITS; ++c) {
    const float ratio = 0.5f + 0.3f * sinf(qw[c * 64]);       // qw[c,0,0]
    const float fc    = ratio * (1.f - ratio);                // sqrt(r)*sqrt(1-r) squared
    float s[8], w[8];
    float mx = -1e30f;
    #pragma unroll
    for (int j = 0; j < 8; ++j) { s[j] = fc * sb[h * 8 + j]; mx = fmaxf(mx, s[j]); }
    float z = 0.f, ssum = 0.f;
    #pragma unroll
    for (int j = 0; j < 8; ++j) { ssum += s[j]; w[j] = expf(s[j] - mx); z += w[j]; }
    const float cp = cosf(atan2f(0.f, ssum));                 // angle of real sum
    #pragma unroll
    for (int j = 0; j < 8; ++j) w[j] = sqrtf(w[j] / z + 1e-12f) * cp;

    #pragma unroll
    for (int i4 = 0; i4 < 4; ++i4) {
      const int d = lane + 32 * i4;
      float wr = 0.f, wi = 0.f;
      #pragma unroll
      for (int j = 0; j < 8; ++j) { wr += w[j] * vre[j][d]; wi += w[j] * vim[j][d]; }
      const float cr = cohr[d], ci = cohi[d];
      const float cor = wr * cr - wi * ci;
      const float coi = wr * ci + wi * cr;
      const float mag = sqrtf(cor * cor + coi * coi);
      float outr;
      if (mag > 1e-30f) {
        const float sc = fminf(fmaxf(mag, 1e-6f), 1.f) / mag; // amplitude clamp, phase kept
        outr = cor * sc;
      } else {
        outr = 1e-6f;                                         // amp=1e-6 at angle 0
      }
      accf[i4] += sup[c] * outr;
    }
  }

  #pragma unroll
  for (int i4 = 0; i4 < 4; ++i4) {
    const int d = lane + 32 * i4;
    measured[(size_t)t * EMBED_DIM + h * HEAD_DIM + d] = accf[i4];
  }
}

// ---------------------------------------------------------------------------
// Launch: GEMM(qkv) -> middle -> GEMM(out)
// ---------------------------------------------------------------------------
extern "C" void kernel_launch(void* const* d_in, const int* in_sizes, int n_in,
                              void* d_out, int out_size, void* d_ws, size_t ws_size,
                              hipStream_t stream) {
  const float* query  = (const float*)d_in[0];  // [2,2048,1024]
  const float* W_qkv  = (const float*)d_in[1];  // [3072,1024]
  const float* b_qkv  = (const float*)d_in[2];  // [3072]
  const float* W_out  = (const float*)d_in[3];  // [1024,1024]
  const float* b_out  = (const float*)d_in[4];  // [1024]
  const float* qw     = (const float*)d_in[5];  // [4,8,8]
  const float* phases = (const float*)d_in[6];  // [16,1024]
  const float* gates  = (const float*)d_in[7];  // [3,8,4]
  const float* logits = (const float*)d_in[8];  // [4]
  float* out = (float*)d_out;

  const int M  = 2 * 2048;       // 4096 tokens
  const int E  = EMBED_DIM;      // 1024
  const int N1 = 3 * EMBED_DIM;  // 3072

  float* qkv      = (float*)d_ws;                 // 4096*3072 fp32 = 48 MB
  float* measured = qkv + (size_t)M * N1;         // 4096*1024 fp32 = 16 MB

  dim3 blk(256);
  wmma_gemm_bf16x3_async<<<dim3(N1 / TN, M / TM), blk, 0, stream>>>(
      query, W_qkv, b_qkv, qkv, M, N1, E);
  quantum_middle<<<dim3(M), blk, 0, stream>>>(
      qkv, phases, gates, qw, logits, measured);
  wmma_gemm_bf16x3_async<<<dim3(E / TN, M / TM), blk, 0, stream>>>(
      measured, W_out, b_out, out, M, E, E);
}
